// MultiHeadedAttention_63556926046965
// MI455X (gfx1250) — matrix operand order repaired and checked by reading, not compiled, not run on hardware
//
#include <hip/hip_runtime.h>
#include <hip/hip_bf16.h>
#include <math.h>

#define HH 12
#define DM 768
#define DK 64
#define BBATCH 2
#define SS 2048
#define NEGVAL -1000000000.0f

typedef __attribute__((ext_vector_type(16))) __bf16 v16bf;
typedef __attribute__((ext_vector_type(8)))  float  v8f;
typedef unsigned int u32x4 __attribute__((ext_vector_type(4)));

union Frag { u32x4 q[2]; v16bf v; };

__device__ inline unsigned short f2bf(float f) {
    unsigned int u = __float_as_uint(f);
    u += 0x7FFFu + ((u >> 16) & 1u);
    return (unsigned short)(u >> 16);
}

// ---------------------------------------------------------------- conversion
__global__ void cvt_f32_to_bf16(const float* __restrict__ s,
                                unsigned short* __restrict__ d, int n) {
    int i = blockIdx.x * blockDim.x + threadIdx.x;
    if (i < n) d[i] = f2bf(s[i]);
}

// ---------------------------------------------------------------- GEMM: Y = X @ W^T + b
// Wave tile: 32(M) x 64(N), 8 C-fragments; A frags reused across 4 N-tiles,
// B frags reused across 2 M-tiles. X: bf16 [M,K] rm. W: bf16 [N,K] rm.
// MODE 0: bf16 head-major out[((b*H+h)*S+s)*64+d]      (Q,K)
// MODE 1: bf16 head-transposed out[((b*H+h)*64+d)*S+s] (V)
// MODE 2: f32 row-major out[m*N+n]                     (final proj)
template <int MODE>
__global__ void __launch_bounds__(128)
gemm_wmma_bf16(const unsigned short* __restrict__ X,
               const unsigned short* __restrict__ W,
               const float* __restrict__ bias,
               void* __restrict__ out,
               int M, int N, int K) {
    const int lane = threadIdx.x & 31;
    const int wave = threadIdx.x >> 5;
    const int half = lane >> 4;
    const int lm   = lane & 15;

    const int ntile = N >> 6;                       // 64-wide N strips
    const int tid = blockIdx.x * 4 + wave;
    const int m0 = (tid / ntile) << 5;              // 32-tall M strips
    const int n0 = (tid % ntile) << 6;

    const unsigned short* xrow0 = X + (size_t)(m0 + lm) * K;
    const unsigned short* xrow1 = X + (size_t)(m0 + 16 + lm) * K;
    const unsigned short* wr[4];
#pragma unroll
    for (int t = 0; t < 4; t++) wr[t] = W + (size_t)(n0 + 16 * t + lm) * K;

    v8f c[2][4];
#pragma unroll
    for (int mt = 0; mt < 2; mt++)
#pragma unroll
        for (int t = 0; t < 4; t++) c[mt][t] = (v8f){};

    for (int kb = 0; kb < K; kb += 32) {
        Frag a0, a1;
        // A 16x32: element e -> K = kb + 16*(e>>3) + 8*half + (e&7)
        a0.q[0] = *(const u32x4*)(xrow0 + kb + 8 * half);
        a0.q[1] = *(const u32x4*)(xrow0 + kb + 16 + 8 * half);
        a1.q[0] = *(const u32x4*)(xrow1 + kb + 8 * half);
        a1.q[1] = *(const u32x4*)(xrow1 + kb + 16 + 8 * half);
        if (kb + 32 < K) {
            __builtin_prefetch(xrow0 + kb + 32);     // global_prefetch_b8
            __builtin_prefetch(xrow1 + kb + 32);
            __builtin_prefetch(wr[0] + kb + 32);
            __builtin_prefetch(wr[2] + kb + 32);
        }
#pragma unroll
        for (int t = 0; t < 4; t++) {
            Frag b;                                  // B: K = kb + 16*(e>>3) + 8*half + (e&7)
            const unsigned short* bp = wr[t] + kb + 8 * half;
            b.q[0] = *(const u32x4*)(bp);
            b.q[1] = *(const u32x4*)(bp + 16);
            c[0][t] = __builtin_amdgcn_wmma_f32_16x16x32_bf16(false, a0.v, false, b.v,
                                                              (short)0, c[0][t], false, false);
            c[1][t] = __builtin_amdgcn_wmma_f32_16x16x32_bf16(false, a1.v, false, b.v,
                                                              (short)0, c[1][t], false, false);
        }
    }

#pragma unroll
    for (int t = 0; t < 4; t++) {
        const int n = n0 + 16 * t + lm;
        const float bn = bias[n];
        const int h = n >> 6, d = n & 63;
#pragma unroll
        for (int mt = 0; mt < 2; mt++) {
#pragma unroll
            for (int r = 0; r < 8; r++) {
                const int m = m0 + 16 * mt + r + 8 * half;
                const float v = c[mt][t][r] + bn;
                if (MODE == 2) {
                    ((float*)out)[(size_t)m * N + n] = v;
                } else {
                    const int bb = m >> 11, s = m & (SS - 1);
                    size_t idx = (MODE == 0)
                        ? (((size_t)(bb * HH + h) * SS + s) * DK + d)
                        : (((size_t)(bb * HH + h) * DK + d) * SS + s);
                    ((unsigned short*)out)[idx] = f2bf(v);
                }
            }
        }
    }
}

// ---------------------------------------------------------------- flash attention
// Q,K: bf16 [B,H,S,DK]; Vt: bf16 [B,H,DK,S]; mask: int [B,S]; XA: bf16 [B,S,DM]
__global__ void __launch_bounds__(128)
attention_wmma(const unsigned short* __restrict__ Q,
               const unsigned short* __restrict__ Km,
               const unsigned short* __restrict__ Vt,
               const int* __restrict__ mask,
               unsigned short* __restrict__ XA) {
    __shared__ __align__(16) unsigned short Plds[4][16 * 32];
    __shared__ __align__(16) unsigned short Vlds[4][64 * 32];

    const int lane = threadIdx.x & 31;
    const int wave = threadIdx.x >> 5;
    const int half = lane >> 4;
    const int lm   = lane & 15;

    const int tid = blockIdx.x * 4 + wave;
    const int b   = tid / (HH * (SS / 16));
    const int rem = tid % (HH * (SS / 16));
    const int h   = rem / (SS / 16);
    const int q0  = (rem % (SS / 16)) << 4;
    const size_t bh = (size_t)(b * HH + h);

    // Q A-fragments for d in [0,32) and [32,64): loaded once
    const unsigned short* qrow = Q + (bh * SS + q0 + lm) * DK;
    Frag qa0, qa1;
    qa0.q[0] = *(const u32x4*)(qrow + 16 * half);
    qa0.q[1] = *(const u32x4*)(qrow + 16 * half + 8);
    qa1.q[0] = *(const u32x4*)(qrow + 32 + 16 * half);
    qa1.q[1] = *(const u32x4*)(qrow + 32 + 16 * half + 8);

    v8f o0 = {}, o1 = {}, o2 = {}, o3 = {};
    float m8[8], l8[8];
#pragma unroll
    for (int r = 0; r < 8; r++) { m8[r] = -INFINITY; l8[r] = 0.f; }

    unsigned short* pl = &Plds[wave][0];
    unsigned short* vl = &Vlds[wave][0];
    const unsigned vl_off = (unsigned)(uintptr_t)vl;   // LDS byte offset (low 32b of aperture addr)

    for (int j0 = 0; j0 < SS; j0 += 32) {
        // ---- async-stage V chunk: Vt rows [0,64) x keys [j0,j0+32) -> Vlds[64][32]
        // WAR: previous chunk's ds reads must land before async writes overwrite LDS
        asm volatile("s_wait_dscnt 0x0" ::: "memory");
#pragma unroll
        for (int i = 0; i < 8; i++) {
            const int idx = i * 32 + lane;             // 16-byte units, 256 total
            const int d = idx >> 2, cseg = idx & 3;    // 4 x 16B per 64B d-row
            const unsigned short* src = Vt + (bh * DK + d) * SS + j0 + cseg * 8;
            const unsigned dst = vl_off + (unsigned)idx * 16u;
            asm volatile("global_load_async_to_lds_b128 %0, %1, off"
                         :: "v"(dst), "v"(src) : "memory");
        }

        // ---- scores S[16 x 32] as two 16x16 f32 fragments (overlaps async copy)
        v8f c0 = {}, c1 = {};
        {
            const unsigned short* krow = Km + (bh * SS + j0 + lm) * DK;
            Frag k0, k1;
            k0.q[0] = *(const u32x4*)(krow + 16 * half);
            k0.q[1] = *(const u32x4*)(krow + 16 * half + 8);
            k1.q[0] = *(const u32x4*)(krow + 32 + 16 * half);
            k1.q[1] = *(const u32x4*)(krow + 32 + 16 * half + 8);
            c0 = __builtin_amdgcn_wmma_f32_16x16x32_bf16(false, qa0.v, false, k0.v,
                                                         (short)0, c0, false, false);
            c0 = __builtin_amdgcn_wmma_f32_16x16x32_bf16(false, qa1.v, false, k1.v,
                                                         (short)0, c0, false, false);
        }
        {
            const unsigned short* krow = Km + (bh * SS + j0 + 16 + lm) * DK;
            Frag k0, k1;
            k0.q[0] = *(const u32x4*)(krow + 16 * half);
            k0.q[1] = *(const u32x4*)(krow + 16 * half + 8);
            k1.q[0] = *(const u32x4*)(krow + 32 + 16 * half);
            k1.q[1] = *(const u32x4*)(krow + 32 + 16 * half + 8);
            c1 = __builtin_amdgcn_wmma_f32_16x16x32_bf16(false, qa0.v, false, k0.v,
                                                         (short)0, c1, false, false);
            c1 = __builtin_amdgcn_wmma_f32_16x16x32_bf16(false, qa1.v, false, k1.v,
                                                         (short)0, c1, false, false);
        }
        // ---- scale + mask (column = j0 + lm / j0 + 16 + lm)
        const int mv0 = mask[b * SS + j0 + lm];
        const int mv1 = mask[b * SS + j0 + 16 + lm];
#pragma unroll
        for (int r = 0; r < 8; r++) {
            c0[r] = (mv0 == 0) ? NEGVAL : c0[r] * 0.125f;
            c1[r] = (mv1 == 0) ? NEGVAL : c1[r] * 0.125f;
        }

        __syncthreads();  // WAR vs previous iteration's P reads
        // ---- online softmax per row (row = r + 8*half; reduce over 16-lane group)
#pragma unroll
        for (int r = 0; r < 8; r++) {
            float mx = fmaxf(c0[r], c1[r]);
            mx = fmaxf(mx, __shfl_xor(mx, 1));
            mx = fmaxf(mx, __shfl_xor(mx, 2));
            mx = fmaxf(mx, __shfl_xor(mx, 4));
            mx = fmaxf(mx, __shfl_xor(mx, 8));
            float nm    = fmaxf(m8[r], mx);
            float alpha = __expf(m8[r] - nm);
            float p0 = __expf(c0[r] - nm);
            float p1 = __expf(c1[r] - nm);
            float rs = p0 + p1;
            rs += __shfl_xor(rs, 1);
            rs += __shfl_xor(rs, 2);
            rs += __shfl_xor(rs, 4);
            rs += __shfl_xor(rs, 8);
            l8[r] = l8[r] * alpha + rs;
            m8[r] = nm;
            o0[r] *= alpha; o1[r] *= alpha; o2[r] *= alpha; o3[r] *= alpha;
            const int row = r + 8 * half;
            pl[row * 32 + lm]      = f2bf(p0);
            pl[row * 32 + 16 + lm] = f2bf(p1);
        }
        __syncthreads();  // P visible across lanes

        // ---- O += P(16x32) @ V(32x64), V fragments from LDS staging
        Frag pa;
        pa.q[0] = *(const u32x4*)(pl + lm * 32 + 16 * half);
        pa.q[1] = *(const u32x4*)(pl + lm * 32 + 16 * half + 8);

        asm volatile("s_wait_asynccnt 0x0" ::: "memory");  // V chunk landed in LDS
        {
            Frag vb;
            vb.q[0] = *(const u32x4*)(vl + (0 + lm) * 32 + 16 * half);
            vb.q[1] = *(const u32x4*)(vl + (0 + lm) * 32 + 16 * half + 8);
            o0 = __builtin_amdgcn_wmma_f32_16x16x32_bf16(false, pa.v, false, vb.v,
                                                         (short)0, o0, false, false);
        }
        {
            Frag vb;
            vb.q[0] = *(const u32x4*)(vl + (16 + lm) * 32 + 16 * half);
            vb.q[1] = *(const u32x4*)(vl + (16 + lm) * 32 + 16 * half + 8);
            o1 = __builtin_amdgcn_wmma_f32_16x16x32_bf16(false, pa.v, false, vb.v,
                                                         (short)0, o1, false, false);
        }
        {
            Frag vb;
            vb.q[0] = *(const u32x4*)(vl + (32 + lm) * 32 + 16 * half);
            vb.q[1] = *(const u32x4*)(vl + (32 + lm) * 32 + 16 * half + 8);
            o2 = __builtin_amdgcn_wmma_f32_16x16x32_bf16(false, pa.v, false, vb.v,
                                                         (short)0, o2, false, false);
        }
        {
            Frag vb;
            vb.q[0] = *(const u32x4*)(vl + (48 + lm) * 32 + 16 * half);
            vb.q[1] = *(const u32x4*)(vl + (48 + lm) * 32 + 16 * half + 8);
            o3 = __builtin_amdgcn_wmma_f32_16x16x32_bf16(false, pa.v, false, vb.v,
                                                         (short)0, o3, false, false);
        }
    }

    // ---- normalize and store XA[b, s, h*64 + d] as bf16
#pragma unroll
    for (int r = 0; r < 8; r++) {
        const float rc = 1.f / l8[r];
        const int s = q0 + r + 8 * half;
        unsigned short* base = XA + ((size_t)b * SS + s) * DM + h * DK;
        base[lm]      = f2bf(o0[r] * rc);
        base[16 + lm] = f2bf(o1[r] * rc);
        base[32 + lm] = f2bf(o2[r] * rc);
        base[48 + lm] = f2bf(o3[r] * rc);
    }
}

// ---------------------------------------------------------------- launcher
extern "C" void kernel_launch(void* const* d_in, const int* in_sizes, int n_in,
                              void* d_out, int out_size, void* d_ws, size_t ws_size,
                              hipStream_t stream) {
    const float* query = (const float*)d_in[0];
    const float* key   = (const float*)d_in[1];
    const float* value = (const float*)d_in[2];
    const int*   mask  = (const int*)d_in[3];
    const float* Wq = (const float*)d_in[4];
    const float* bq = (const float*)d_in[5];
    const float* Wk = (const float*)d_in[6];
    const float* bk = (const float*)d_in[7];
    const float* Wv = (const float*)d_in[8];
    const float* bv = (const float*)d_in[9];
    const float* Wo = (const float*)d_in[10];
    const float* bo = (const float*)d_in[11];
    float* out = (float*)d_out;

    // workspace layout (bf16 halves), ~46.5 MiB total
    const size_t NX = (size_t)BBATCH * SS * DM;  // 3,145,728
    const size_t NW = (size_t)DM * DM;           //   589,824
    unsigned short* w  = (unsigned short*)d_ws;
    unsigned short* XQ = w;  w += NX;
    unsigned short* XK = w;  w += NX;
    unsigned short* XV = w;  w += NX;
    unsigned short* WQb = w; w += NW;
    unsigned short* WKb = w; w += NW;
    unsigned short* WVb = w; w += NW;
    unsigned short* WOb = w; w += NW;
    unsigned short* QH = w;  w += NX;   // [B,H,S,DK]
    unsigned short* KH = w;  w += NX;   // [B,H,S,DK]
    unsigned short* VT = w;  w += NX;   // [B,H,DK,S]
    unsigned short* XA = w;  w += NX;   // [B,S,DM]

    // fp32 -> bf16 conversions
    cvt_f32_to_bf16<<<(int)(NX / 256), 256, 0, stream>>>(query, XQ, (int)NX);
    cvt_f32_to_bf16<<<(int)(NX / 256), 256, 0, stream>>>(key,   XK, (int)NX);
    cvt_f32_to_bf16<<<(int)(NX / 256), 256, 0, stream>>>(value, XV, (int)NX);
    cvt_f32_to_bf16<<<(int)(NW / 256), 256, 0, stream>>>(Wq, WQb, (int)NW);
    cvt_f32_to_bf16<<<(int)(NW / 256), 256, 0, stream>>>(Wk, WKb, (int)NW);
    cvt_f32_to_bf16<<<(int)(NW / 256), 256, 0, stream>>>(Wv, WVb, (int)NW);
    cvt_f32_to_bf16<<<(int)(NW / 256), 256, 0, stream>>>(Wo, WOb, (int)NW);

    const int M = BBATCH * SS, N = DM, K = DM;
    const int tiles = (M / 32) * (N / 64);       // 1536 wave-tasks -> 384 blocks
    dim3 blk(128);

    gemm_wmma_bf16<0><<<tiles / 4, blk, 0, stream>>>(XQ, WQb, bq, (void*)QH, M, N, K);
    gemm_wmma_bf16<0><<<tiles / 4, blk, 0, stream>>>(XK, WKb, bk, (void*)KH, M, N, K);
    gemm_wmma_bf16<1><<<tiles / 4, blk, 0, stream>>>(XV, WVb, bv, (void*)VT, M, N, K);

    const int qtasks = BBATCH * HH * (SS / 16);  // 3072 waves -> 768 blocks
    attention_wmma<<<qtasks / 4, blk, 0, stream>>>(QH, KH, VT, mask, XA);

    gemm_wmma_bf16<2><<<tiles / 4, blk, 0, stream>>>(XA, WOb, bo, (void*)out, M, N, K);
}
